// NormalSelfAttention_1726576856580
// MI455X (gfx1250) — matrix operand order repaired and checked by reading, not compiled, not run on hardware
//
#include <hip/hip_runtime.h>
#include <hip/hip_bf16.h>
#include <math.h>

// ---------------------------------------------------------------------------
// Problem constants (from the reference)
// ---------------------------------------------------------------------------
#define BB     4
#define LL     1360          // 1024 + 256 + 64 + 16
#define DM     1024
#define NH     16
#define DK     64
#define MROWS  (BB * LL)     // 5440

typedef __bf16 v8bf  __attribute__((ext_vector_type(8)));
typedef __bf16 v16bf __attribute__((ext_vector_type(16)));
typedef float  v8f   __attribute__((ext_vector_type(8)));

// ---------------------------------------------------------------------------
// WMMA helpers (CDNA5 gfx1250, wave32)
// ---------------------------------------------------------------------------
__device__ __forceinline__ v8f wmma_bf16(v16bf a, v16bf b, v8f c) {
  // D = A(16x32 bf16) * B(32x16 bf16) + C(16x16 f32)
  return __builtin_amdgcn_wmma_f32_16x16x32_bf16(
      /*neg_a=*/false, a, /*neg_b=*/false, b,
      /*c_mod=*/(short)0, c, /*reuse_a=*/false, /*reuse_b=*/false);
}

// A fragment (16x32, 16-bit): lane half h -> K in {8h..8h+7} then {16+8h..23+8h}
__device__ __forceinline__ v16bf load_a_frag(const __bf16* p, int half) {
  v8bf lo = *reinterpret_cast<const v8bf*>(p + 8 * half);
  v8bf hi = *reinterpret_cast<const v8bf*>(p + 16 + 8 * half);
  return __builtin_shufflevector(lo, hi, 0,1,2,3,4,5,6,7,8,9,10,11,12,13,14,15);
}

// B fragment (32x16, 16-bit): lane half h -> K in {8h..8h+7} then {16+8h..23+8h}
__device__ __forceinline__ v16bf load_b_frag(const __bf16* p, int half) {
  v8bf lo = *reinterpret_cast<const v8bf*>(p + 8 * half);
  v8bf hi = *reinterpret_cast<const v8bf*>(p + 16 + 8 * half);
  return __builtin_shufflevector(lo, hi, 0,1,2,3,4,5,6,7,8,9,10,11,12,13,14,15);
}

// cross-lane reductions within each 16-lane column group (wave32)
__device__ __forceinline__ float redmax16(float v) {
  #pragma unroll
  for (int m = 1; m < 16; m <<= 1) v = fmaxf(v, __shfl_xor(v, m, 16));
  return v;
}
__device__ __forceinline__ float redsum16(float v) {
  #pragma unroll
  for (int m = 1; m < 16; m <<= 1) v += __shfl_xor(v, m, 16);
  return v;
}

// ---------------------------------------------------------------------------
// Analytic PAM mask.  Levels: [0,1024) [1024,1280) [1280,1344) [1344,1360)
// allowed(i,j): same level and |i-j|<=2, or parent<->child (window 4) link.
// ---------------------------------------------------------------------------
__device__ __forceinline__ int pam_level(int i) {
  return (i < 1024) ? 0 : (i < 1280) ? 1 : (i < 1344) ? 2 : 3;
}
__device__ __forceinline__ int pam_start(int lvl) {
  return (lvl == 0) ? 0 : (lvl == 1) ? 1024 : (lvl == 2) ? 1280 : 1344;
}
__device__ __forceinline__ bool pam_allowed(int i, int j) {
  int li = pam_level(i), lj = pam_level(j);
  if (li == lj) { int d = i - j; return d <= 2 && d >= -2; }
  if (lj == li + 1) return (j - pam_start(lj)) == ((i - pam_start(li)) >> 2);
  if (li == lj + 1) return (i - pam_start(li)) == ((j - pam_start(lj)) >> 2);
  return false;
}

// ---------------------------------------------------------------------------
// Kernel 0: weight convert + transpose  W[K=1024][N=1024] f32 -> WT[N][K] bf16
// ---------------------------------------------------------------------------
__global__ void k_wt(const float* __restrict__ W, __bf16* __restrict__ WT) {
  __shared__ float tile[32][33];
  int n0 = blockIdx.x * 32, k0 = blockIdx.y * 32;
  int tx = threadIdx.x, ty = threadIdx.y;                // 32 x 8
  #pragma unroll
  for (int i = 0; i < 32; i += 8)
    tile[ty + i][tx] = W[(size_t)(k0 + ty + i) * DM + (n0 + tx)];
  __syncthreads();
  #pragma unroll
  for (int i = 0; i < 32; i += 8)
    WT[(size_t)(n0 + ty + i) * DM + (k0 + tx)] = (__bf16)tile[tx][ty + i];
}

// ---------------------------------------------------------------------------
// Kernel 1: LayerNorm -> qin (bf16) and plain bf16 copy of x (GEMM A operands)
// one 256-thread block per row of 1024
// ---------------------------------------------------------------------------
__global__ void k_ln(const float* __restrict__ x, const float* __restrict__ gamma,
                     const float* __restrict__ beta, __bf16* __restrict__ qin,
                     __bf16* __restrict__ xb) {
  __shared__ float red[256];
  int row = blockIdx.x, t = threadIdx.x;
  const float* xr = x + (size_t)row * DM;
  float4 v = reinterpret_cast<const float4*>(xr)[t];
  red[t] = v.x + v.y + v.z + v.w;
  __syncthreads();
  #pragma unroll
  for (int o = 128; o > 0; o >>= 1) {
    if (t < o) red[t] += red[t + o];
    __syncthreads();
  }
  float mu = red[0] * (1.0f / DM);
  __syncthreads();
  float dx0 = v.x - mu, dx1 = v.y - mu, dx2 = v.z - mu, dx3 = v.w - mu;
  red[t] = dx0 * dx0 + dx1 * dx1 + dx2 * dx2 + dx3 * dx3;
  __syncthreads();
  #pragma unroll
  for (int o = 128; o > 0; o >>= 1) {
    if (t < o) red[t] += red[t + o];
    __syncthreads();
  }
  float rstd = rsqrtf(red[0] * (1.0f / DM) + 1e-6f);
  int c = t * 4;
  size_t base = (size_t)row * DM + c;
  qin[base + 0] = (__bf16)(dx0 * rstd * gamma[c + 0] + beta[c + 0]);
  qin[base + 1] = (__bf16)(dx1 * rstd * gamma[c + 1] + beta[c + 1]);
  qin[base + 2] = (__bf16)(dx2 * rstd * gamma[c + 2] + beta[c + 2]);
  qin[base + 3] = (__bf16)(dx3 * rstd * gamma[c + 3] + beta[c + 3]);
  xb[base + 0] = (__bf16)v.x;  xb[base + 1] = (__bf16)v.y;
  xb[base + 2] = (__bf16)v.z;  xb[base + 3] = (__bf16)v.w;
}

// ---------------------------------------------------------------------------
// Kernel 2: bf16 WMMA GEMM  C[M x 1024] = alpha * A[M x 1024] * WT^T
//   one wave per 32x64 C tile; K-loop step 32, register double-buffered:
//   next iteration's 6 fragments are loaded before this iteration's 8 WMMAs.
//   __launch_bounds__(32, 1): allow the full wave32 VGPR budget so all
//   fragments + 8 accumulators stay resident (no scratch spills).
//   mode 0: bf16 row-major out          (Q, K projections)
//   mode 1: bf16 transposed V out:  vt[((b*16+h)*64+d)*L + s]
//   mode 2: f32 out + bias + residual   (final FC)
// ---------------------------------------------------------------------------
__global__ void __launch_bounds__(32, 1)
k_gemm(const __bf16* __restrict__ A, const __bf16* __restrict__ WT,
       void* __restrict__ Cout, const float* __restrict__ bias,
       const float* __restrict__ resid, float alpha, int mode) {
  int mt = blockIdx.x;                 // 170 M-tiles of 32
  int n0 = blockIdx.y * 64;            // 16 N-tiles of 64
  int lane = threadIdx.x;
  int lm = lane & 15, half = lane >> 4;

  const __bf16* ar0 = A + (size_t)(mt * 32 + lm) * DM;        // rows 0..15
  const __bf16* ar1 = A + (size_t)(mt * 32 + 16 + lm) * DM;   // rows 16..31
  const __bf16* br0 = WT + (size_t)(n0 + 0 * 16 + lm) * DM;
  const __bf16* br1 = WT + (size_t)(n0 + 1 * 16 + lm) * DM;
  const __bf16* br2 = WT + (size_t)(n0 + 2 * 16 + lm) * DM;
  const __bf16* br3 = WT + (size_t)(n0 + 3 * 16 + lm) * DM;

  v8f acc00 = {}, acc01 = {}, acc02 = {}, acc03 = {};
  v8f acc10 = {}, acc11 = {}, acc12 = {}, acc13 = {};

  v16bf a0 = load_a_frag(ar0, half);
  v16bf a1 = load_a_frag(ar1, half);
  v16bf b0 = load_b_frag(br0, half);
  v16bf b1 = load_b_frag(br1, half);
  v16bf b2 = load_b_frag(br2, half);
  v16bf b3 = load_b_frag(br3, half);

  for (int k0 = 32; k0 < DM; k0 += 32) {
    // prefetch next k-step's fragments first (hidden behind the WMMAs below)
    v16bf na0 = load_a_frag(ar0 + k0, half);
    v16bf na1 = load_a_frag(ar1 + k0, half);
    v16bf nb0 = load_b_frag(br0 + k0, half);
    v16bf nb1 = load_b_frag(br1 + k0, half);
    v16bf nb2 = load_b_frag(br2 + k0, half);
    v16bf nb3 = load_b_frag(br3 + k0, half);
    acc00 = wmma_bf16(a0, b0, acc00);
    acc01 = wmma_bf16(a0, b1, acc01);
    acc02 = wmma_bf16(a0, b2, acc02);
    acc03 = wmma_bf16(a0, b3, acc03);
    acc10 = wmma_bf16(a1, b0, acc10);
    acc11 = wmma_bf16(a1, b1, acc11);
    acc12 = wmma_bf16(a1, b2, acc12);
    acc13 = wmma_bf16(a1, b3, acc13);
    a0 = na0; a1 = na1; b0 = nb0; b1 = nb1; b2 = nb2; b3 = nb3;
  }
  // tail k-step
  acc00 = wmma_bf16(a0, b0, acc00);
  acc01 = wmma_bf16(a0, b1, acc01);
  acc02 = wmma_bf16(a0, b2, acc02);
  acc03 = wmma_bf16(a0, b3, acc03);
  acc10 = wmma_bf16(a1, b0, acc10);
  acc11 = wmma_bf16(a1, b1, acc11);
  acc12 = wmma_bf16(a1, b2, acc12);
  acc13 = wmma_bf16(a1, b3, acc13);

  const v8f accs[2][4] = {{acc00, acc01, acc02, acc03},
                          {acc10, acc11, acc12, acc13}};
  #pragma unroll
  for (int mi = 0; mi < 2; ++mi) {
    #pragma unroll
    for (int nt = 0; nt < 4; ++nt) {
      #pragma unroll
      for (int r = 0; r < 8; ++r) {
        int row = mt * 32 + mi * 16 + r + 8 * half;
        int col = n0 + nt * 16 + lm;
        float val = accs[mi][nt][r] * alpha;
        if (mode == 0) {
          ((__bf16*)Cout)[(size_t)row * DM + col] = (__bf16)val;
        } else if (mode == 1) {
          int b = row / LL, s = row % LL;
          int h = col >> 6, d = col & 63;
          ((__bf16*)Cout)[((size_t)((b * NH + h) * DK + d)) * LL + s] = (__bf16)val;
        } else {
          ((float*)Cout)[(size_t)row * DM + col] =
              val + bias[col] + resid[(size_t)row * DM + col];
        }
      }
    }
  }
}

// ---------------------------------------------------------------------------
// Kernel 3: flash attention, one wave per (b, h, 16-row q tile)
//   Q,K: bf16 [B, L, H, DK]; VT: bf16 [B, H, DK, L]; CTX: bf16 [B*L, 1024]
//   V fragments are loaded at the top of each chunk so their latency hides
//   behind the score WMMAs and the softmax VALU work.
// ---------------------------------------------------------------------------
__global__ void __launch_bounds__(32, 1)
k_attn(const __bf16* __restrict__ Q, const __bf16* __restrict__ K,
       const __bf16* __restrict__ VT, __bf16* __restrict__ CTX) {
  __shared__ __bf16 ldsP[16 * 32];     // P tile, C-layout -> A-layout staging
  int qt = blockIdx.x, h = blockIdx.y, b = blockIdx.z;
  int lane = threadIdx.x;
  int lm = lane & 15, half = lane >> 4;

  // Q A-fragments for this 16-row tile (K dim = dk, split 0..31 / 32..63)
  const __bf16* qrow = Q + ((size_t)(b * LL + qt * 16 + lm) * NH + h) * DK;
  v16bf qa0 = load_a_frag(qrow + 0, half);
  v16bf qa1 = load_a_frag(qrow + 32, half);

  v8f acc0 = {}, acc1 = {}, acc2 = {}, acc3 = {};
  float mrow[8], lrow[8];
  #pragma unroll
  for (int r = 0; r < 8; ++r) { mrow[r] = -1e30f; lrow[r] = 0.0f; }

  const __bf16* vbase = VT + (size_t)((b * NH + h) * DK) * LL;

  for (int kc = 0; kc < LL; kc += 32) {
    // ---- issue V fragment loads early (consumed only at chunk bottom) ----
    v16bf vb0 = load_b_frag(vbase + (size_t)(0 * 16 + lm) * LL + kc, half);
    v16bf vb1 = load_b_frag(vbase + (size_t)(1 * 16 + lm) * LL + kc, half);
    v16bf vb2 = load_b_frag(vbase + (size_t)(2 * 16 + lm) * LL + kc, half);
    v16bf vb3 = load_b_frag(vbase + (size_t)(3 * 16 + lm) * LL + kc, half);

    // ---- scores: two 16x16 tiles, K(dk)=64 via two chained WMMAs each ----
    int kr1 = kc + 16 + lm; if (kr1 >= LL) kr1 = LL - 1;   // clamp tail reads
    const __bf16* k0row = K + ((size_t)(b * LL + kc + lm) * NH + h) * DK;
    const __bf16* k1row = K + ((size_t)(b * LL + kr1) * NH + h) * DK;
    v16bf kb00 = load_b_frag(k0row + 0, half);
    v16bf kb01 = load_b_frag(k0row + 32, half);
    v16bf kb10 = load_b_frag(k1row + 0, half);
    v16bf kb11 = load_b_frag(k1row + 32, half);
    v8f s0 = {}, s1 = {};
    s0 = wmma_bf16(qa0, kb00, s0);
    s0 = wmma_bf16(qa1, kb01, s0);
    s1 = wmma_bf16(qa0, kb10, s1);
    s1 = wmma_bf16(qa1, kb11, s1);

    // ---- mask + online softmax (rows r+8*half live in this lane) ----
    int j0 = kc + lm, j1 = kc + 16 + lm;
    int iBase = qt * 16 + 8 * half;
    float sc[8];
    #pragma unroll
    for (int r = 0; r < 8; ++r) {
      int i = iBase + r;
      float a0 = (pam_allowed(i, j0)) ? s0[r] : -1e9f;              // j0 < LL
      float a1 = (j1 < LL && pam_allowed(i, j1)) ? s1[r] : -1e9f;
      float cmax = redmax16(fmaxf(a0, a1));
      float mnew = fmaxf(mrow[r], cmax);
      float p0 = __expf(a0 - mnew);
      float p1 = __expf(a1 - mnew);
      float rs = redsum16(p0 + p1);
      float s = __expf(mrow[r] - mnew);
      lrow[r] = lrow[r] * s + rs;
      mrow[r] = mnew;
      sc[r] = s;
      ldsP[(r + 8 * half) * 32 + lm]      = (__bf16)p0;
      ldsP[(r + 8 * half) * 32 + 16 + lm] = (__bf16)p1;
    }
    #pragma unroll
    for (int r = 0; r < 8; ++r) {
      acc0[r] *= sc[r]; acc1[r] *= sc[r]; acc2[r] *= sc[r]; acc3[r] *= sc[r];
    }
    __syncthreads();   // P tile visible in A-layout order

    // ---- ctx += P(16x32) x V(32x64): 4 WMMAs against transposed V ----
    v16bf pa = load_a_frag(&ldsP[lm * 32], half);
    acc0 = wmma_bf16(pa, vb0, acc0);
    acc1 = wmma_bf16(pa, vb1, acc1);
    acc2 = wmma_bf16(pa, vb2, acc2);
    acc3 = wmma_bf16(pa, vb3, acc3);
    __syncthreads();   // done with ldsP before next chunk overwrites
  }

  // ---- normalize and store ctx (bf16, row-major [B*L, H*DK]) ----
  #pragma unroll
  for (int r = 0; r < 8; ++r) {
    float inv = 1.0f / lrow[r];
    size_t grow = (size_t)(b * LL + qt * 16 + r + 8 * half) * DM + h * DK + lm;
    CTX[grow + 0]  = (__bf16)(acc0[r] * inv);
    CTX[grow + 16] = (__bf16)(acc1[r] * inv);
    CTX[grow + 32] = (__bf16)(acc2[r] * inv);
    CTX[grow + 48] = (__bf16)(acc3[r] * inv);
  }
}

// ---------------------------------------------------------------------------
// Launch: x, Wq, Wk, Wv, Wfc, bfc, gamma, beta, mask(unused; mask is analytic)
// ---------------------------------------------------------------------------
extern "C" void kernel_launch(void* const* d_in, const int* in_sizes, int n_in,
                              void* d_out, int out_size, void* d_ws, size_t ws_size,
                              hipStream_t stream) {
  const float* x     = (const float*)d_in[0];
  const float* Wq    = (const float*)d_in[1];
  const float* Wk    = (const float*)d_in[2];
  const float* Wv    = (const float*)d_in[3];
  const float* Wfc   = (const float*)d_in[4];
  const float* bfc   = (const float*)d_in[5];
  const float* gamma = (const float*)d_in[6];
  const float* beta  = (const float*)d_in[7];
  // d_in[8] (bool mask) intentionally unused: the PAM mask is computed analytically.

  char* ws = (char*)d_ws;
  const size_t WT_SZ  = (size_t)DM * DM * 2;          // 2 MB per bf16 weight
  const size_t ACT_SZ = (size_t)MROWS * DM * 2;       // 11.14 MB per bf16 activation
  __bf16* wqT  = (__bf16*)(ws + 0 * WT_SZ);
  __bf16* wkT  = (__bf16*)(ws + 1 * WT_SZ);
  __bf16* wvT  = (__bf16*)(ws + 2 * WT_SZ);
  __bf16* wfcT = (__bf16*)(ws + 3 * WT_SZ);
  char*   act  = ws + 4 * WT_SZ;
  __bf16* qin  = (__bf16*)(act + 0 * ACT_SZ);   // reused as CTX after Q-GEMM
  __bf16* xb   = (__bf16*)(act + 1 * ACT_SZ);
  __bf16* qb   = (__bf16*)(act + 2 * ACT_SZ);
  __bf16* kb   = (__bf16*)(act + 3 * ACT_SZ);
  __bf16* vt   = (__bf16*)(act + 4 * ACT_SZ);   // V transposed [B,H,DK,L]
  __bf16* ctxb = qin;
  (void)in_sizes; (void)n_in; (void)out_size; (void)ws_size;

  dim3 wtg(DM / 32, DM / 32), wtb(32, 8);
  k_wt<<<wtg, wtb, 0, stream>>>(Wq, wqT);
  k_wt<<<wtg, wtb, 0, stream>>>(Wk, wkT);
  k_wt<<<wtg, wtb, 0, stream>>>(Wv, wvT);
  k_wt<<<wtg, wtb, 0, stream>>>(Wfc, wfcT);

  k_ln<<<MROWS, 256, 0, stream>>>(x, gamma, beta, qin, xb);

  dim3 gg(MROWS / 32, DM / 64);
  k_gemm<<<gg, 32, 0, stream>>>(qin, wqT, (void*)qb, nullptr, nullptr, 0.125f, 0); // /sqrt(64)
  k_gemm<<<gg, 32, 0, stream>>>(xb, wkT, (void*)kb, nullptr, nullptr, 1.0f, 0);
  k_gemm<<<gg, 32, 0, stream>>>(xb, wvT, (void*)vt, nullptr, nullptr, 1.0f, 1);

  k_attn<<<dim3(LL / 16, NH, BB), 32, 0, stream>>>(qb, kb, vt, ctxb);

  k_gemm<<<gg, 32, 0, stream>>>(ctxb, wfcT, d_out, bfc, x, 1.0f, 2);
}
